// LinformerAttention_77283641524843
// MI455X (gfx1250) — matrix operand order repaired and checked by reading, not compiled, not run on hardware
//
#include <hip/hip_runtime.h>

// ---------------------------------------------------------------------------
// Linformer attention for MI455X (gfx1250): all GEMMs via v_wmma_f32_16x16x32_f16,
// attention stages Kp into LDS via the Tensor Data Mover (TENSORcnt tracked).
// ---------------------------------------------------------------------------

typedef __attribute__((ext_vector_type(16))) _Float16     v16h;
typedef __attribute__((ext_vector_type(8)))  _Float16     v8h;
typedef __attribute__((ext_vector_type(8)))  float        v8f;
typedef __attribute__((ext_vector_type(4)))  unsigned int v4u;
typedef __attribute__((ext_vector_type(4)))  int          v4i;
typedef __attribute__((ext_vector_type(8)))  int          v8i;

#define DEV __device__ __forceinline__

constexpr int BB = 4;
constexpr int T  = 4096;
constexpr int D  = 1024;
constexpr int H  = 16;
constexpr int HD = 64;
constexpr int LR = 256;
constexpr int BT = BB * T;   // 16384
constexpr int N3 = 3 * D;    // 3072

#if defined(__gfx1250__) && __has_builtin(__builtin_amdgcn_tensor_load_to_lds)
#define HAVE_TDM 1
#else
#define HAVE_TDM 0
#endif

// ---- WMMA fragment helpers (wave32) ---------------------------------------
// A (16x32 f16): lane = half*16 + rf; row = rf; holds K = half*8 + {0..7} and
// half*8 + 16 + {0..7}  -> two 16B loads at +0 / +16 elements.
DEV v16h load_a_frag(const _Float16* p) {
  union { v16h v; v8h h[2]; } u;
  u.h[0] = *(const v8h*)(p);
  u.h[1] = *(const v8h*)(p + 16);
  return u.v;
}
// B (32x16 f16) from the TRANSPOSED matrix Bt (N x K row-major):
// lane column n = rf; holds K = half*8 + {0..7} and half*8 + 16 + {0..7}
// -> two 16B loads at +0 / +16 elements.
DEV v16h load_b_frag(const _Float16* p) {
  union { v16h v; v8h h[2]; } u;
  u.h[0] = *(const v8h*)(p);
  u.h[1] = *(const v8h*)(p + 16);
  return u.v;
}
DEV v8f wmma_f16(v16h a, v16h b, v8f c) {
  return __builtin_amdgcn_wmma_f32_16x16x32_f16(false, a, false, b,
                                                (short)0, c, false, false);
}
DEV v8f vzero() { v8f z = {0.f,0.f,0.f,0.f,0.f,0.f,0.f,0.f}; return z; }

// ---- Tensor Data Mover: contiguous global -> LDS copy ----------------------
// Packs a minimal D# (count=1, type=2 "image", data_size=8B, 1-row tile of
// nbytes/8 elements) per CDNA5 ISA and issues
// TENSOR_LOAD_TO_LDS. Tracked by TENSORcnt. nbytes must be a multiple of 8
// and nbytes/8 must fit in tile_dim0's 16 bits (<= 512KB copies).
DEV void tdm_copy_1d(void* lds_ptr, const void* gptr, unsigned nbytes) {
#if HAVE_TDM
  const unsigned lds = (unsigned)(size_t)lds_ptr;
  const unsigned long long ga = (unsigned long long)(size_t)gptr;
  const unsigned n8 = nbytes >> 3;            // elements of 8 bytes
  v4u g0;
  g0.x = 1u;                                  // count=1, user mode, no gather
  g0.y = lds;                                 // D#.lds_addr
  g0.z = (unsigned)ga;                        // D#.global_addr[31:0]
  g0.w = (unsigned)(ga >> 32) | (2u << 30);   // global_addr[56:32] | type=2
  v8i g1;
  g1[0] = (int)(3u << 16);                    // wg_mask=0, data_size=3 (8B)
  g1[1] = (int)((n8 & 0xFFFFu) << 16);        // tensor_dim0[15:0]  (bits 63:48)
  g1[2] = (int)((n8 >> 16) | (1u << 16));     // tensor_dim0[31:16], tensor_dim1=1
  g1[3] = (int)((n8 & 0xFFFFu) << 16);        // tile_dim0 = n8     (bits 127:112)
  g1[4] = 1;                                  // tile_dim1 = 1, tile_dim2 = 0
  g1[5] = (int)n8;                            // tensor_dim0_stride[31:0]
  g1[6] = 0;
  g1[7] = 0;
  v4i z4 = {0, 0, 0, 0};
#if defined(__clang_major__) && __clang_major__ >= 23
  v8i z8 = {0, 0, 0, 0, 0, 0, 0, 0};
  __builtin_amdgcn_tensor_load_to_lds(g0, g1, z4, z4, z8, 0);
#else
  __builtin_amdgcn_tensor_load_to_lds(g0, g1, z4, z4, 0);
#endif
#else
  (void)lds_ptr; (void)gptr; (void)nbytes;
#endif
}

// ---------------------------------------------------------------------------
// K0: one-shot layout/precision conversion.
//   Wt[j][i]  = f16(W_kernel[i][j])        (3072 x 1024)
//   Ot[j][i]  = f16(o_kernel[i][j])        (1024 x 1024)
//   Et[h][k][t] = f16(E[t][h][k])          (16 x 256 x 4096)
//   Ft[h][k][t] = f16(F[t][h][k])
// ---------------------------------------------------------------------------
__global__ __launch_bounds__(256)
void convert_pack(const float* __restrict__ Wk, const float* __restrict__ Ok,
                  const float* __restrict__ E,  const float* __restrict__ F,
                  _Float16* __restrict__ Wt, _Float16* __restrict__ Ot,
                  _Float16* __restrict__ Et, _Float16* __restrict__ Ft) {
  const long nW = (long)D * N3;
  const long nO = (long)D * D;
  const long nE = (long)T * H * LR;
  const long total = nW + nO + 2 * nE;
  for (long idx = (long)blockIdx.x * blockDim.x + threadIdx.x; idx < total;
       idx += (long)gridDim.x * blockDim.x) {
    long q = idx;
    if (q < nW) {
      long j = q / D, i = q - j * D;
      Wt[q] = (_Float16)Wk[i * N3 + j];
    } else if ((q -= nW) < nO) {
      long j = q / D, i = q - j * D;
      Ot[q] = (_Float16)Ok[i * D + j];
    } else if ((q -= nO) < nE) {
      long h = q / ((long)LR * T), r = q - h * (long)LR * T;
      long k = r / T, t = r - k * T;
      Et[q] = (_Float16)E[(t * H + h) * LR + k];
    } else {
      q -= nE;
      long h = q / ((long)LR * T), r = q - h * (long)LR * T;
      long k = r / T, t = r - k * T;
      Ft[q] = (_Float16)F[(t * H + h) * LR + k];
    }
  }
}

// ---------------------------------------------------------------------------
// K1: per-row L2 normalization, f32 -> f16.  One 256-thread block per row.
// ---------------------------------------------------------------------------
__global__ __launch_bounds__(256)
void rownorm(const float* __restrict__ X, _Float16* __restrict__ Xn) {
  const long row = blockIdx.x;
  const float* x = X + row * (long)D;
  float s = 0.f;
  for (int i = threadIdx.x; i < D; i += 256) { float v = x[i]; s += v * v; }
  #pragma unroll
  for (int off = 16; off > 0; off >>= 1) s += __shfl_xor(s, off, 32);
  __shared__ float wsum[8];
  __shared__ float sinv;
  if ((threadIdx.x & 31) == 0) wsum[threadIdx.x >> 5] = s;
  __syncthreads();
  if (threadIdx.x == 0) {
    float t = 0.f;
    #pragma unroll
    for (int i = 0; i < 8; ++i) t += wsum[i];
    sinv = rsqrtf(t);
  }
  __syncthreads();
  const float r = sinv;
  _Float16* o = Xn + row * (long)D;
  for (int i = threadIdx.x; i < D; i += 256) o[i] = (_Float16)(x[i] * r);
}

// ---------------------------------------------------------------------------
// K2: QKV GEMM.  C(16384x3072) = Xn @ W, bias added; scattered into
//   Q  [B*H, T, HD]  (row-major "A" layout for the logits GEMM)
//   Kt [B*H, HD, T]  (transposed "B" layout for the low-rank GEMM)
//   Vt [B*H, HD, T]
// One wave per block; each wave computes a 32x64 tile (2x4 WMMA accumulators).
// ---------------------------------------------------------------------------
__global__ __launch_bounds__(32)
void gemm_qkv(const _Float16* __restrict__ Xn, const _Float16* __restrict__ Wt,
              const float* __restrict__ bias,
              _Float16* __restrict__ Qb, _Float16* __restrict__ Kt,
              _Float16* __restrict__ Vt) {
  const int lane = threadIdx.x & 31, rf = lane & 15, half = lane >> 4;
  const int m0 = blockIdx.y * 32, n0 = blockIdx.x * 64;
  v8f acc[2][4];
  #pragma unroll
  for (int a = 0; a < 2; ++a)
    #pragma unroll
    for (int b = 0; b < 4; ++b) acc[a][b] = vzero();

  const _Float16* A0 = Xn + (size_t)(m0 + rf) * D + half * 8;
  const _Float16* A1 = A0 + (size_t)16 * D;
  const _Float16* B0 = Wt + (size_t)(n0 + rf) * D + half * 8;
  for (int k = 0; k < D; k += 32) {
    v16h a0 = load_a_frag(A0 + k);
    v16h a1 = load_a_frag(A1 + k);
    #pragma unroll
    for (int j = 0; j < 4; ++j) {
      v16h b = load_b_frag(B0 + (size_t)j * 16 * D + k);
      acc[0][j] = wmma_f16(a0, b, acc[0][j]);
      acc[1][j] = wmma_f16(a1, b, acc[1][j]);
    }
    __builtin_prefetch(A0 + k + 256, 0, 1);
    __builtin_prefetch(B0 + k + 256, 0, 1);
  }
  #pragma unroll
  for (int mt = 0; mt < 2; ++mt) {
    #pragma unroll
    for (int j = 0; j < 4; ++j) {
      const int col = n0 + j * 16 + rf;         // constant sel/h within tile
      const int sel = col >> 10, r = col & 1023, h = r >> 6, d = r & 63;
      const float bv = bias[col];
      #pragma unroll
      for (int v = 0; v < 8; ++v) {
        const int row = m0 + mt * 16 + v + 8 * half;   // global token index
        const int b_ = row >> 12, t = row & 4095;
        const _Float16 hv = (_Float16)(acc[mt][j][v] + bv);
        const size_t bh = (size_t)(b_ * H + h);
        if (sel == 0)       Qb[(bh * T + t) * HD + d] = hv;
        else if (sel == 1)  Kt[(bh * HD + d) * T + t] = hv;
        else                Vt[(bh * HD + d) * T + t] = hv;
      }
    }
  }
}

// ---------------------------------------------------------------------------
// K3: low-rank projections. Per (b,h):  C(256x64) = Et_h(256x4096) @ K_bh^T.
//   z==0: Kp [bh][k][d] (row-major, becomes Bt for logits GEMM)
//   z==1: Vp stored transposed: Vpt [bh][d][k] (becomes Bt for the PV GEMM)
// Wave tile: 16 x 64 (1x4 accumulators), K-loop over T=4096.
// ---------------------------------------------------------------------------
__global__ __launch_bounds__(32)
void gemm_lowrank(const _Float16* __restrict__ Et, const _Float16* __restrict__ Ft,
                  const _Float16* __restrict__ Kt, const _Float16* __restrict__ Vt,
                  _Float16* __restrict__ Kp, _Float16* __restrict__ Vpt) {
  const int lane = threadIdx.x & 31, rf = lane & 15, half = lane >> 4;
  const int bh = blockIdx.y, h = bh & (H - 1);
  const int which = blockIdx.z;
  const int k0 = blockIdx.x * 16;

  const _Float16* A  = (which ? Ft : Et) + ((size_t)h * LR + k0 + rf) * T + half * 8;
  const _Float16* Bt = (which ? Vt : Kt) + ((size_t)bh * HD + rf) * T + half * 8;

  v8f acc[4];
  #pragma unroll
  for (int j = 0; j < 4; ++j) acc[j] = vzero();

  for (int t = 0; t < T; t += 32) {
    v16h a = load_a_frag(A + t);
    #pragma unroll
    for (int j = 0; j < 4; ++j) {
      v16h b = load_b_frag(Bt + (size_t)j * 16 * T + t);
      acc[j] = wmma_f16(a, b, acc[j]);
    }
    __builtin_prefetch(A + t + 256, 0, 1);
    __builtin_prefetch(Bt + t + 256, 0, 1);
  }
  if (which == 0) {
    _Float16* o = Kp + (size_t)bh * LR * HD;
    #pragma unroll
    for (int j = 0; j < 4; ++j)
      #pragma unroll
      for (int v = 0; v < 8; ++v)
        o[(size_t)(k0 + v + 8 * half) * HD + j * 16 + rf] = (_Float16)acc[j][v];
  } else {
    _Float16* o = Vpt + (size_t)bh * HD * LR;
    #pragma unroll
    for (int j = 0; j < 4; ++j)
      #pragma unroll
      for (int v = 0; v < 8; ++v)
        o[(size_t)(j * 16 + rf) * LR + k0 + v + 8 * half] = (_Float16)acc[j][v];
  }
}

// ---------------------------------------------------------------------------
// K4: fused attention. 4 waves / block; block handles 64 T-rows of one head.
//   Kp (256x64 f16 = 32KB) is staged into LDS ONCE per block by the Tensor
//   Data Mover (wave 0 issues; s_wait_tensorcnt + barrier), then all 4 waves
//   read their logits B-fragments from LDS -> 4x less L2 traffic.
//   logits(16x256) = Q(16x64) @ Kp^T   [16 WMMA accumulators / wave]
//   softmax over 256 via __shfl_xor row reductions
//   probs -> LDS (C-layout -> A-layout reshape), then
//   ctx(16x64) = P(16x256) @ Vp        [Bt = Vpt from global/L2]
// ---------------------------------------------------------------------------
constexpr int AW = 4;  // waves per attention block

__global__ __launch_bounds__(32 * AW)
void attention(const _Float16* __restrict__ Qb, const _Float16* __restrict__ Kp,
               const _Float16* __restrict__ Vpt, _Float16* __restrict__ ctx) {
  __shared__ __align__(16) _Float16 sKp[LR * HD];      // 32 KB
  __shared__ __align__(16) _Float16 sP[AW][16 * LR];   // 4 x 8 KB
  const int tid = threadIdx.x;
  const int wave = tid >> 5, lane = tid & 31, rf = lane & 15, half = lane >> 4;
  const int bh = blockIdx.y;
  const int t0 = blockIdx.x * (16 * AW) + wave * 16;

  const _Float16* Kpb = Kp  + (size_t)bh * LR * HD;
  const _Float16* Vpb = Vpt + (size_t)bh * HD * LR;

#if HAVE_TDM
  if (wave == 0) {
    tdm_copy_1d(&sKp[0], Kpb, LR * HD * sizeof(_Float16));
    __builtin_amdgcn_s_wait_tensorcnt(0);
  }
  __syncthreads();
#else
  {
    const v8h* src = (const v8h*)Kpb;
    v8h* dst = (v8h*)sKp;
    for (int i = tid; i < LR * HD / 8; i += 32 * AW) dst[i] = src[i];
    __syncthreads();
  }
#endif

  const _Float16* Aq = Qb + ((size_t)bh * T + t0 + rf) * HD + half * 8;
  v8f acc[16];
  #pragma unroll
  for (int j = 0; j < 16; ++j) acc[j] = vzero();

  for (int k = 0; k < HD; k += 32) {
    v16h a = load_a_frag(Aq + k);
    #pragma unroll
    for (int j = 0; j < 16; ++j) {
      v16h b = load_b_frag(&sKp[(size_t)(j * 16 + rf) * HD + half * 8 + k]);
      acc[j] = wmma_f16(a, b, acc[j]);
    }
  }

  const float scale = 0.125f;   // 1/sqrt(64)
  _Float16* myP = &sP[wave][0];
  #pragma unroll
  for (int v = 0; v < 8; ++v) {
    const int row = v + 8 * half;
    float mx = -3.0e38f;
    #pragma unroll
    for (int j = 0; j < 16; ++j) mx = fmaxf(mx, acc[j][v]);
    #pragma unroll
    for (int off = 1; off < 16; off <<= 1) mx = fmaxf(mx, __shfl_xor(mx, off, 32));
    float e[16], sum = 0.f;
    #pragma unroll
    for (int j = 0; j < 16; ++j) { e[j] = __expf((acc[j][v] - mx) * scale); sum += e[j]; }
    #pragma unroll
    for (int off = 1; off < 16; off <<= 1) sum += __shfl_xor(sum, off, 32);
    const float inv = 1.0f / sum;
    #pragma unroll
    for (int j = 0; j < 16; ++j)
      myP[row * LR + j * 16 + rf] = (_Float16)(e[j] * inv);
  }
  __syncthreads();

  v8f oac[4];
  #pragma unroll
  for (int j = 0; j < 4; ++j) oac[j] = vzero();
  const _Float16* Ap = &myP[rf * LR + half * 8];
  for (int k = 0; k < LR; k += 32) {
    v16h a = load_a_frag(Ap + k);
    #pragma unroll
    for (int j = 0; j < 4; ++j) {
      v16h b = load_b_frag(Vpb + (size_t)(j * 16 + rf) * LR + half * 8 + k);
      oac[j] = wmma_f16(a, b, oac[j]);
    }
  }
  const int b_ = bh >> 4, h = bh & 15;
  _Float16* o = ctx + ((size_t)b_ * T + t0) * D + h * HD;
  #pragma unroll
  for (int j = 0; j < 4; ++j)
    #pragma unroll
    for (int v = 0; v < 8; ++v)
      o[(size_t)(v + 8 * half) * D + j * 16 + rf] = (_Float16)oac[j][v];
}

// ---------------------------------------------------------------------------
// K5: output GEMM.  out(16384x1024) = ctx @ o_kernel + o_bias, f32 output.
// Same 32x64 wave tiling as K2; Bt = o_kernel^T.
// ---------------------------------------------------------------------------
__global__ __launch_bounds__(32)
void gemm_out(const _Float16* __restrict__ ctx, const _Float16* __restrict__ Ot,
              const float* __restrict__ ob, float* __restrict__ out) {
  const int lane = threadIdx.x & 31, rf = lane & 15, half = lane >> 4;
  const int m0 = blockIdx.y * 32, n0 = blockIdx.x * 64;
  v8f acc[2][4];
  #pragma unroll
  for (int a = 0; a < 2; ++a)
    #pragma unroll
    for (int b = 0; b < 4; ++b) acc[a][b] = vzero();

  const _Float16* A0 = ctx + (size_t)(m0 + rf) * D + half * 8;
  const _Float16* A1 = A0 + (size_t)16 * D;
  const _Float16* B0 = Ot + (size_t)(n0 + rf) * D + half * 8;
  for (int k = 0; k < D; k += 32) {
    v16h a0 = load_a_frag(A0 + k);
    v16h a1 = load_a_frag(A1 + k);
    #pragma unroll
    for (int j = 0; j < 4; ++j) {
      v16h b = load_b_frag(B0 + (size_t)j * 16 * D + k);
      acc[0][j] = wmma_f16(a0, b, acc[0][j]);
      acc[1][j] = wmma_f16(a1, b, acc[1][j]);
    }
    __builtin_prefetch(A0 + k + 256, 0, 1);
    __builtin_prefetch(B0 + k + 256, 0, 1);
  }
  #pragma unroll
  for (int mt = 0; mt < 2; ++mt)
    #pragma unroll
    for (int j = 0; j < 4; ++j) {
      const int col = n0 + j * 16 + rf;
      const float bv = ob[col];
      #pragma unroll
      for (int v = 0; v < 8; ++v) {
        const int row = m0 + mt * 16 + v + 8 * half;
        out[(size_t)row * D + col] = acc[mt][j][v] + bv;
      }
    }
}

// ---------------------------------------------------------------------------
// Workspace layout (bytes). Total ~214 MB; every operand of every GEMM fits in
// the 192 MB L2 individually, which is what the single-wave-tile schedule
// leans on.
// ---------------------------------------------------------------------------
constexpr size_t OFF_XN  = 0;                       // 16384*1024 f16
constexpr size_t OFF_WT  = 33554432;                // 3072*1024  f16
constexpr size_t OFF_OT  = 39845888;                // 1024*1024  f16
constexpr size_t OFF_ET  = 41943040;                // 16*256*4096 f16
constexpr size_t OFF_FT  = 58720256;
constexpr size_t OFF_QB  = 75497472;                // 64*4096*64 f16
constexpr size_t OFF_KT  = 109051904;               // 64*64*4096 f16
constexpr size_t OFF_VT  = 142606336;
constexpr size_t OFF_KP  = 176160768;               // 64*256*64 f16
constexpr size_t OFF_VP  = 178257920;
constexpr size_t OFF_CTX = 180355072;               // 16384*1024 f16

extern "C" void kernel_launch(void* const* d_in, const int* in_sizes, int n_in,
                              void* d_out, int out_size, void* d_ws, size_t ws_size,
                              hipStream_t stream) {
  (void)in_sizes; (void)n_in; (void)out_size; (void)ws_size;
  const float* X  = (const float*)d_in[0];
  const float* Wk = (const float*)d_in[1];
  const float* Wb = (const float*)d_in[2];
  const float* E  = (const float*)d_in[3];
  const float* F  = (const float*)d_in[4];
  const float* Ok = (const float*)d_in[5];
  const float* Ob = (const float*)d_in[6];
  float* out = (float*)d_out;

  char* ws = (char*)d_ws;
  _Float16* Xn  = (_Float16*)(ws + OFF_XN);
  _Float16* Wt  = (_Float16*)(ws + OFF_WT);
  _Float16* Ot  = (_Float16*)(ws + OFF_OT);
  _Float16* Et  = (_Float16*)(ws + OFF_ET);
  _Float16* Ft  = (_Float16*)(ws + OFF_FT);
  _Float16* Qb  = (_Float16*)(ws + OFF_QB);
  _Float16* Kt  = (_Float16*)(ws + OFF_KT);
  _Float16* Vt  = (_Float16*)(ws + OFF_VT);
  _Float16* Kp  = (_Float16*)(ws + OFF_KP);
  _Float16* Vpt = (_Float16*)(ws + OFF_VP);
  _Float16* Ctx = (_Float16*)(ws + OFF_CTX);

  convert_pack<<<dim3(8192), dim3(256), 0, stream>>>(Wk, Ok, E, F, Wt, Ot, Et, Ft);
  rownorm<<<dim3(BT), dim3(256), 0, stream>>>(X, Xn);
  gemm_qkv<<<dim3(N3 / 64, BT / 32), dim3(32), 0, stream>>>(Xn, Wt, Wb, Qb, Kt, Vt);
  gemm_lowrank<<<dim3(LR / 16, BB * H, 2), dim3(32), 0, stream>>>(Et, Ft, Kt, Vt, Kp, Vpt);
  attention<<<dim3(T / (16 * AW), BB * H), dim3(32 * AW), 0, stream>>>(Qb, Kp, Vpt, Ctx);
  gemm_out<<<dim3(D / 64, BT / 32), dim3(32), 0, stream>>>(Ctx, Ot, Ob, out);
}
